// FastAttention_52398601011709
// MI455X (gfx1250) — matrix operand order repaired and checked by reading, not compiled, not run on hardware
//
#include <hip/hip_runtime.h>
#include <hip/hip_bf16.h>
#include <math.h>

// ---------------- problem constants ----------------
#define BH   16          // B*H
#define NTOK 16384       // sequence length
#define DD   64          // head dim
#define MREAL 266        // random features
#define MP   288         // padded M: 18 tiles of 16 = 9 K-chunks of 32
#define MT   18
#define KC   9
#define NWAVE 9
#define TPB  288

#define DN     0.35355339059327373f   // 64^-0.25
#define DN2H   0.0625f                // 0.5 * DN^2
#define RATIO  0.06131393394849658f   // 266^-0.5
#define EPSF   1e-4f

typedef __attribute__((ext_vector_type(16))) __bf16 v16bf;
typedef __attribute__((ext_vector_type(8)))  __bf16 v8bf;
typedef __attribute__((ext_vector_type(8)))  float  v8f;

// ---------------- workspace layout (bytes) ----------------
static constexpr size_t O_PROJW = 0;                       // bf16 [MP][DD]
static constexpr size_t O_CTXTB = 36864;                   // bf16 [BH][DD][MP]
static constexpr size_t O_STAB  = 626688;                  // f32  [BH]
static constexpr size_t O_KSUM  = 626944;                  // f32  [BH][MP]
static constexpr size_t O_CTXT  = 645376;                  // f32  [BH][DD][MP]
static constexpr size_t WS_NEED = 1825024;

// ---------------- helpers ----------------
static __device__ __forceinline__ v8f wmma_bf16(v16bf a, v16bf b, v8f c) {
  // v_wmma_f32_16x16x32_bf16: D = A(16x32) * B(32x16) + C(16x16 f32)
  return __builtin_amdgcn_wmma_f32_16x16x32_bf16(false, a, false, b, (short)0, c,
                                                 false, false);
}

struct AFrag { v16bf a; float ss; };

// A-matrix fragment (16x32 bf16) from fp32 rows [row..row+15] of x (ld = DD),
// scaled by DN; also accumulates per-lane sum of squares of raw x.
// Lane layout: lane L (g=L/16, m=L&15): element e -> K=kbase+g*16+e.
static __device__ __forceinline__ AFrag load_afrag(const float* __restrict__ xr,
                                                   int kbase, int lane) {
  int g = lane >> 4, mr = lane & 15;
  const float* p = xr + mr * DD;
  AFrag f; f.ss = 0.f;
#pragma unroll
  for (int e = 0; e < 8; e++) {
    float x = p[kbase + g * 16 + e];
    f.ss = fmaf(x, x, f.ss);
    f.a[e] = (__bf16)(x * DN);
  }
#pragma unroll
  for (int e = 0; e < 8; e++) {
    float x = p[kbase + g * 16 + 8 + e];
    f.ss = fmaf(x, x, f.ss);
    f.a[e + 8] = (__bf16)(x * DN);
  }
  return f;
}

// A-matrix fragment of v^T (rows = d, K = 32 sequence rows), unscaled.
static __device__ __forceinline__ v16bf load_vt(const float* __restrict__ v0,
                                                int dbase, int lane) {
  int g = lane >> 4, dr = dbase + (lane & 15);
  v16bf a;
#pragma unroll
  for (int e = 0; e < 8; e++) a[e] = (__bf16)v0[(g * 16 + e) * DD + dr];
#pragma unroll
  for (int e = 0; e < 8; e++) a[e + 8] = (__bf16)v0[(g * 16 + 8 + e) * DD + dr];
  return a;
}

static __device__ __forceinline__ void atomicMaxF(float* addr, float v) {
  unsigned* ua = (unsigned*)addr;
  unsigned cur = __float_as_uint(-__builtin_inff());
  while (__uint_as_float(cur) < v) {
    unsigned assumed = cur;
    cur = atomicCAS(ua, assumed, __float_as_uint(v));
    if (cur == assumed) break;
  }
}

// ---------------- kernel 0: init + projection fp32 -> bf16 (padded) ----------
__global__ void perf_init_kernel(const float* __restrict__ proj,
                                 __bf16* __restrict__ projW,
                                 float* __restrict__ stab,
                                 float* __restrict__ ksum,
                                 float* __restrict__ ctxT) {
  int i = blockIdx.x * blockDim.x + threadIdx.x;
  int stride = gridDim.x * blockDim.x;
  for (int t = i; t < MP * DD; t += stride) {
    int m = t / DD, d = t % DD;
    projW[t] = (__bf16)((m < MREAL) ? proj[m * DD + d] : 0.f);
  }
  for (int t = i; t < BH; t += stride) stab[t] = -__builtin_inff();
  for (int t = i; t < BH * MP; t += stride) ksum[t] = 0.f;
  for (size_t t = i; t < (size_t)BH * DD * MP; t += stride) ctxT[t] = 0.f;
}

// ---------------- kernel 1: global max of data_dash_k per head --------------
__global__ void __launch_bounds__(TPB)
perf_kmax_kernel(const float* __restrict__ kk, const __bf16* __restrict__ projW,
                 float* __restrict__ stab) {
  int bh = blockIdx.y;
  int lane = threadIdx.x & 31, w = threadIdx.x >> 5;
  int g = lane >> 4, mr = lane & 15;
  const float* kh = kk + (size_t)bh * NTOK * DD;
  bool ok[2] = { (w * 2) * 16 + mr < MREAL, (w * 2 + 1) * 16 + mr < MREAL };
  float mx = -__builtin_inff();
  for (int tile = blockIdx.x; tile < NTOK / 16; tile += gridDim.x) {
    const float* xr = kh + (size_t)tile * 16 * DD;
    int nt = tile + gridDim.x;
    if (nt < NTOK / 16)
      __builtin_prefetch((const char*)(kh + (size_t)nt * 16 * DD) + threadIdx.x * 16, 0, 3);
    AFrag A0 = load_afrag(xr, 0, lane);
    AFrag A1 = load_afrag(xr, 32, lane);
#pragma unroll
    for (int t = 0; t < 2; t++) {
      int mt = w * 2 + t;
      v16bf B0 = *(const v16bf*)(projW + (mt * 16 + mr) * DD + g * 16);
      v16bf B1 = *(const v16bf*)(projW + (mt * 16 + mr) * DD + 32 + g * 16);
      v8f c = {0.f, 0.f, 0.f, 0.f, 0.f, 0.f, 0.f, 0.f};
      c = wmma_bf16(A0.a, B0, c);
      c = wmma_bf16(A1.a, B1, c);
#pragma unroll
      for (int r = 0; r < 8; r++) mx = fmaxf(mx, ok[t] ? c[r] : -__builtin_inff());
    }
  }
#pragma unroll
  for (int s = 16; s > 0; s >>= 1) mx = fmaxf(mx, __shfl_xor(mx, s, 32));
  if (lane == 0) atomicMaxF(&stab[bh], mx);
}

// ---------------- kernel 2: kp, k_cumsum, context = kp^T v ------------------
__global__ void __launch_bounds__(TPB)
perf_kpctx_kernel(const float* __restrict__ kk, const float* __restrict__ vv,
                  const __bf16* __restrict__ projW, const float* __restrict__ stab,
                  float* __restrict__ ksum, float* __restrict__ ctxT) {
  __shared__ __bf16 kpT[NWAVE * 32 * 32];  // per-wave [32 m][32 n] bf16
  int bh = blockIdx.y;
  int lane = threadIdx.x & 31, w = threadIdx.x >> 5;
  int g = lane >> 4, mr = lane & 15;
  const float stabv = stab[bh];
  const float* kh = kk + (size_t)bh * NTOK * DD;
  const float* vh = vv + (size_t)bh * NTOK * DD;
  __bf16* myKp = kpT + w * 32 * 32;

  v8f ctxc[2][4];
#pragma unroll
  for (int t = 0; t < 2; t++)
#pragma unroll
    for (int j = 0; j < 4; j++) ctxc[t][j] = (v8f){0.f,0.f,0.f,0.f,0.f,0.f,0.f,0.f};
  float kcum[2] = {0.f, 0.f};

  for (int tile = blockIdx.x; tile < NTOK / 32; tile += gridDim.x) {
    int nbase = tile * 32;
    int nt = tile + gridDim.x;
    if (nt < NTOK / 32) {
      __builtin_prefetch((const char*)(kh + (size_t)nt * 32 * DD) + threadIdx.x * 32, 0, 3);
      __builtin_prefetch((const char*)(vh + (size_t)nt * 32 * DD) + threadIdx.x * 32, 0, 3);
    }
    // ---- phase 1: kp for both 16-row halves into LDS (own m-columns) ----
#pragma unroll
    for (int h = 0; h < 2; h++) {
      const float* xr = kh + (size_t)(nbase + h * 16) * DD;
      AFrag A0 = load_afrag(xr, 0, lane);
      AFrag A1 = load_afrag(xr, 32, lane);
      float ssq = A0.ss + A1.ss;
      ssq += __shfl_xor(ssq, 16, 32);          // full 64-k sumsq for row lane&15
      float diagv = DN2H * ssq;
      float dvals[8];
#pragma unroll
      for (int r = 0; r < 8; r++) dvals[r] = __shfl(diagv, r + 8 * g, 32);
#pragma unroll
      for (int t = 0; t < 2; t++) {
        int mt = w * 2 + t;
        v16bf B0 = *(const v16bf*)(projW + (mt * 16 + mr) * DD + g * 16);
        v16bf B1 = *(const v16bf*)(projW + (mt * 16 + mr) * DD + 32 + g * 16);
        v8f c = {0.f, 0.f, 0.f, 0.f, 0.f, 0.f, 0.f, 0.f};
        c = wmma_bf16(A0.a, B0, c);
        c = wmma_bf16(A1.a, B1, c);
        bool ok = (mt * 16 + mr) < MREAL;
        v8bf kpv;
        float rsum = 0.f;
#pragma unroll
        for (int r = 0; r < 8; r++) {
          float kp = ok ? RATIO * (__expf(c[r] - dvals[r] - stabv) + EPSF) : 0.f;
          rsum += kp;
          kpv[r] = (__bf16)kp;
        }
        kcum[t] += rsum;
        *(v8bf*)(myKp + (t * 16 + mr) * 32 + h * 16 + 8 * g) = kpv;
      }
    }
    // ---- phase 2: context += v^T(16d x 32n) * kp(32n x 16m), K = 32 ----
#pragma unroll
    for (int j = 0; j < 4; j++) {
      v16bf Av = load_vt(vh + (size_t)nbase * DD, j * 16, lane);
#pragma unroll
      for (int t = 0; t < 2; t++) {
        v16bf Bk = *(const v16bf*)(myKp + (t * 16 + mr) * 32 + g * 16);
        ctxc[t][j] = wmma_bf16(Av, Bk, ctxc[t][j]);
      }
    }
  }
  // ---- flush k_cumsum ----
#pragma unroll
  for (int t = 0; t < 2; t++) {
    float s = kcum[t] + __shfl_xor(kcum[t], 16, 32);
    if (lane < 16) atomicAdd(&ksum[bh * MP + (w * 2 + t) * 16 + mr], s);
  }
  // ---- flush context (fp32 accumulators, transposed [d][m]) ----
#pragma unroll
  for (int t = 0; t < 2; t++)
#pragma unroll
    for (int j = 0; j < 4; j++)
#pragma unroll
      for (int r = 0; r < 8; r++) {
        int d = j * 16 + r + 8 * g;
        int m = (w * 2 + t) * 16 + mr;
        atomicAdd(&ctxT[((size_t)bh * DD + d) * MP + m], ctxc[t][j][r]);
      }
}

// ---------------- kernel 2b: context fp32 -> bf16 ---------------------------
__global__ void perf_ctxcvt_kernel(const float* __restrict__ ctxT,
                                   __bf16* __restrict__ ctxTb) {
  size_t stride = (size_t)gridDim.x * blockDim.x;
  for (size_t i = blockIdx.x * (size_t)blockDim.x + threadIdx.x;
       i < (size_t)BH * DD * MP; i += stride)
    ctxTb[i] = (__bf16)ctxT[i];
}

// ---------------- kernel 3: qp, d_inv, out = qp * context * d_inv -----------
__global__ void __launch_bounds__(TPB)
perf_qout_kernel(const float* __restrict__ qq, const __bf16* __restrict__ projW,
                 const float* __restrict__ ksum, const __bf16* __restrict__ ctxTb,
                 float* __restrict__ out) {
  __shared__ __bf16 qp[32 * MP];             // [row][m] bf16
  __shared__ float rowmaxp[NWAVE][32];
  __shared__ float rowmax[32];
  __shared__ float dinvp[32];
  int bh = blockIdx.y;
  int lane = threadIdx.x & 31, w = threadIdx.x >> 5;
  int g = lane >> 4, mr = lane & 15;
  const float* qh = qq + (size_t)bh * NTOK * DD;
  const __bf16* ctxb = ctxTb + (size_t)bh * DD * MP;
  bool ok[2] = { (w * 2) * 16 + mr < MREAL, (w * 2 + 1) * 16 + mr < MREAL };

  for (int tile = blockIdx.x; tile < NTOK / 32; tile += gridDim.x) {
    int nbase = tile * 32;
    __syncthreads();                          // qp/dinvp reuse from prev iter
    if (threadIdx.x < 32) dinvp[threadIdx.x] = 0.f;
    int nt = tile + gridDim.x;
    if (nt < NTOK / 32)
      __builtin_prefetch((const char*)(qh + (size_t)nt * 32 * DD) + threadIdx.x * 32, 0, 3);

    // ---- phase A: data_dash for own 2 m-tiles, both halves ----
    v8f ddc[2][2];
    float dval[2][8];
#pragma unroll
    for (int h = 0; h < 2; h++) {
      const float* xr = qh + (size_t)(nbase + h * 16) * DD;
      AFrag A0 = load_afrag(xr, 0, lane);
      AFrag A1 = load_afrag(xr, 32, lane);
      float ssq = A0.ss + A1.ss;
      ssq += __shfl_xor(ssq, 16, 32);
      float diagv = DN2H * ssq;
#pragma unroll
      for (int r = 0; r < 8; r++) dval[h][r] = __shfl(diagv, r + 8 * g, 32);
#pragma unroll
      for (int t = 0; t < 2; t++) {
        int mt = w * 2 + t;
        v16bf B0 = *(const v16bf*)(projW + (mt * 16 + mr) * DD + g * 16);
        v16bf B1 = *(const v16bf*)(projW + (mt * 16 + mr) * DD + 32 + g * 16);
        v8f c = {0.f, 0.f, 0.f, 0.f, 0.f, 0.f, 0.f, 0.f};
        c = wmma_bf16(A0.a, B0, c);
        c = wmma_bf16(A1.a, B1, c);
        ddc[h][t] = c;
      }
    }
    // ---- per-row max partial (over own m) -> LDS ----
#pragma unroll
    for (int h = 0; h < 2; h++)
#pragma unroll
      for (int r = 0; r < 8; r++) {
        float a = ok[0] ? ddc[h][0][r] : -__builtin_inff();
        float b = ok[1] ? ddc[h][1][r] : -__builtin_inff();
        float pm = fmaxf(a, b);
        pm = fmaxf(pm, __shfl_xor(pm, 1, 32));
        pm = fmaxf(pm, __shfl_xor(pm, 2, 32));
        pm = fmaxf(pm, __shfl_xor(pm, 4, 32));
        pm = fmaxf(pm, __shfl_xor(pm, 8, 32));
        if (mr == r) rowmaxp[w][h * 16 + r + 8 * g] = pm;
      }
    __syncthreads();
    if (threadIdx.x < 32) {
      float m = rowmaxp[0][threadIdx.x];
#pragma unroll
      for (int ww = 1; ww < NWAVE; ww++) m = fmaxf(m, rowmaxp[ww][threadIdx.x]);
      rowmax[threadIdx.x] = m;
    }
    __syncthreads();
    // ---- qp into LDS + d_inv partials ----
#pragma unroll
    for (int h = 0; h < 2; h++) {
      float dpart[8] = {0.f, 0.f, 0.f, 0.f, 0.f, 0.f, 0.f, 0.f};
#pragma unroll
      for (int t = 0; t < 2; t++) {
        int m = (w * 2 + t) * 16 + mr;
        float ksv = ok[t] ? ksum[bh * MP + m] : 0.f;
#pragma unroll
        for (int r = 0; r < 8; r++) {
          int row = h * 16 + r + 8 * g;
          float qv = ok[t]
              ? RATIO * (__expf(ddc[h][t][r] - dval[h][r] - rowmax[row]) + EPSF)
              : 0.f;
          qp[row * MP + m] = (__bf16)qv;
          dpart[r] += qv * ksv;
        }
      }
#pragma unroll
      for (int r = 0; r < 8; r++) {
        float s = dpart[r];
        s += __shfl_xor(s, 1, 32);
        s += __shfl_xor(s, 2, 32);
        s += __shfl_xor(s, 4, 32);
        s += __shfl_xor(s, 8, 32);
        if (mr == r) atomicAdd(&dinvp[h * 16 + r + 8 * g], s);
      }
    }
    __syncthreads();
    // ---- phase B: out(16n x 16d) = qp(16n x 288) * ctx(288 x 16d), 9 chunks
    if (w < 8) {
      int h = w >> 2, j = w & 3;
      v8f oc = {0.f, 0.f, 0.f, 0.f, 0.f, 0.f, 0.f, 0.f};
#pragma unroll
      for (int c = 0; c < KC; c++) {
        const __bf16* pr = &qp[(h * 16 + mr) * MP + c * 32];
        v8bf lo = *(const v8bf*)(pr + g * 16);
        v8bf hi = *(const v8bf*)(pr + g * 16 + 8);
        v16bf Aq;
#pragma unroll
        for (int e = 0; e < 8; e++) { Aq[e] = lo[e]; Aq[e + 8] = hi[e]; }
        v16bf Bc = *(const v16bf*)(ctxb + (j * 16 + mr) * MP + c * 32 + g * 16);
        oc = wmma_bf16(Aq, Bc, oc);
      }
#pragma unroll
      for (int r = 0; r < 8; r++) {
        int row = h * 16 + r + 8 * g;
        out[((size_t)bh * NTOK + nbase + row) * DD + j * 16 + mr] =
            oc[r] * (1.0f / dinvp[row]);
      }
    }
  }
}

// ---------------- launch ----------------------------------------------------
extern "C" void kernel_launch(void* const* d_in, const int* in_sizes, int n_in,
                              void* d_out, int out_size, void* d_ws, size_t ws_size,
                              hipStream_t stream) {
  (void)in_sizes; (void)n_in; (void)out_size;
  if (ws_size < WS_NEED) return;
  const float* q    = (const float*)d_in[0];
  const float* k    = (const float*)d_in[1];
  const float* v    = (const float*)d_in[2];
  const float* proj = (const float*)d_in[3];
  float* out = (float*)d_out;
  char* ws = (char*)d_ws;
  __bf16* projW = (__bf16*)(ws + O_PROJW);
  __bf16* ctxTb = (__bf16*)(ws + O_CTXTB);
  float*  stab  = (float*)(ws + O_STAB);
  float*  ksum  = (float*)(ws + O_KSUM);
  float*  ctxT  = (float*)(ws + O_CTXT);

  perf_init_kernel<<<512, 256, 0, stream>>>(proj, projW, stab, ksum, ctxT);
  perf_kmax_kernel<<<dim3(32, BH), TPB, 0, stream>>>(k, projW, stab);
  perf_kpctx_kernel<<<dim3(32, BH), TPB, 0, stream>>>(k, v, projW, stab, ksum, ctxT);
  perf_ctxcvt_kernel<<<288, 256, 0, stream>>>(ctxT, ctxTb);
  perf_qout_kernel<<<dim3(64, BH), TPB, 0, stream>>>(q, projW, ksum, ctxTb, out);
}
